// GRUMappingNoGCN_47373489274934
// MI455X (gfx1250) — matrix operand order repaired and checked by reading, not compiled, not run on hardware
//
#include <hip/hip_runtime.h>
#include <hip/hip_bf16.h>

// Problem constants (fixed by the reference setup_inputs()).
constexpr int B_   = 2;
constexpr int N1_  = 4096;
constexpr int N2_  = 8192;
constexpr int CIN_ = 67;   // C + 3
constexpr int KNN_ = 16;

typedef __attribute__((ext_vector_type(16))) _Float16 v16h;
typedef __attribute__((ext_vector_type(8)))  _Float16 v8h;
typedef __attribute__((ext_vector_type(8)))  float    v8f;
typedef __attribute__((ext_vector_type(4)))  float    v4f;

#define WMMA_F16(acc, afrag, bfrag)                                            \
  acc = __builtin_amdgcn_wmma_f32_16x16x32_f16(false, afrag, false, bfrag,     \
                                               (short)0, acc, false, false)

// f16 weight workspace layout (element offsets), K padded: 96 for CIN, 64 for C
constexpr int WO_R0  = 0;            // [64][96]
constexpr int WO_Z0  = 6144;         // [64][96]
constexpr int WO_H0  = 12288;        // [64][96]
constexpr int WO_R1  = 18432;        // [64][64]
constexpr int WO_Z1  = 22528;
constexpr int WO_H1  = 26624;
constexpr int WO_FR  = 30720;
constexpr int WO_FZ  = 34816;
constexpr int WO_FRO = 38912;
constexpr int WF_TOTAL = 43008;

// ---------------------------------------------------------------------------
// Kernel 0: convert all weight matrices f32 -> f16 (K zero-padded).
// ---------------------------------------------------------------------------
__global__ __launch_bounds__(256) void wcvt_kernel(
    const float* __restrict__ w_r0, const float* __restrict__ w_z0,
    const float* __restrict__ w_h0, const float* __restrict__ w_r1,
    const float* __restrict__ w_z1, const float* __restrict__ w_h1,
    const float* __restrict__ w_fr, const float* __restrict__ w_fz,
    const float* __restrict__ w_fro, _Float16* __restrict__ wf) {
  const int t = blockIdx.x * 256 + threadIdx.x;
  if (t >= WF_TOTAL) return;
  if (t < 18432) {  // three [64][96] matrices (valid K < 67)
    const int wi = t / 6144, r = t % 6144, m = r / 96, k = r % 96;
    const float* src = (wi == 0) ? w_r0 : (wi == 1) ? w_z0 : w_h0;
    wf[t] = (k < CIN_) ? (_Float16)src[m * CIN_ + k] : (_Float16)0.0f;
  } else {          // six [64][64] matrices
    const int t2 = t - 18432, wi = t2 / 4096, r = t2 % 4096;
    const float* src = (wi == 0) ? w_r1 : (wi == 1) ? w_z1 : (wi == 2) ? w_h1
                     : (wi == 3) ? w_fr : (wi == 4) ? w_fz : w_fro;
    wf[t] = (_Float16)src[r];
  }
}

// ---------------------------------------------------------------------------
// Kernel 1: exact KNN (K=16 smallest squared distances), one query per thread,
// xyz2 streamed through LDS in 1024-point tiles. Unrolled register top-16.
// ---------------------------------------------------------------------------
#define TILE2 1024

__global__ __launch_bounds__(256) void knn_kernel(const float* __restrict__ xyz1,
                                                  const float* __restrict__ xyz2,
                                                  int* __restrict__ knn) {
  __shared__ float sx[TILE2], sy[TILE2], sz[TILE2];
  const int b  = blockIdx.y;
  const int n1 = blockIdx.x * 256 + threadIdx.x;
  const float* x1 = xyz1 + (size_t)b * 3 * N1_;
  const float* x2 = xyz2 + (size_t)b * 3 * N2_;
  const float qx = x1[n1], qy = x1[N1_ + n1], qz = x1[2 * N1_ + n1];

  float bd[KNN_];
  int   bi[KNN_];
#pragma unroll
  for (int k = 0; k < KNN_; ++k) { bd[k] = 3.0e38f; bi[k] = 0; }

  for (int base = 0; base < N2_; base += TILE2) {
    for (int i = threadIdx.x; i < TILE2; i += 256) {
      sx[i] = x2[base + i];
      sy[i] = x2[N2_ + base + i];
      sz[i] = x2[2 * N2_ + base + i];
    }
    __syncthreads();
    for (int i = 0; i < TILE2; ++i) {
      const float dx = sx[i] - qx, dy = sy[i] - qy, dz = sz[i] - qz;
      const float d = dx * dx + dy * dy + dz * dz;
      if (d < bd[KNN_ - 1]) {
        bd[KNN_ - 1] = d;
        bi[KNN_ - 1] = base + i;
#pragma unroll
        for (int j = KNN_ - 1; j > 0; --j) {
          if (bd[j] < bd[j - 1]) {
            float td = bd[j]; bd[j] = bd[j - 1]; bd[j - 1] = td;
            int   ti = bi[j]; bi[j] = bi[j - 1]; bi[j - 1] = ti;
          }
        }
      }
    }
    __syncthreads();
  }
  int* o = knn + ((size_t)(b * N1_) + n1) * KNN_;
#pragma unroll
  for (int k = 0; k < KNN_; ++k) o[k] = bi[k];
}

// ---------------------------------------------------------------------------
// Fragment loaders, vectorized (two 16B loads each).
// A (16x32 MxK), f16 weights row-major [M][ldk] (ldk mult of 8, zero-padded):
//   lane L: M = rowBase + (L&15); 16 consecutive K at 32kt + 16*(L>>4).
// B (32x16 KxN), LDS column-major-K [col][ldk]:
//   lane L: N = colBase + (L&15); 16 consecutive K at 32kt + 16*(L>>4).
// C/D f32: elem j -> M = rowBase + j + 8*(L>>4), N = colBase + (L&15):
//   8 consecutive M per lane -> one 16B LDS store per epilogue.
// ---------------------------------------------------------------------------
__device__ __forceinline__ v16h frag_a_wf(const _Float16* __restrict__ Wf, int ldk,
                                          int rowBase, int kt, int lane) {
  const int m = rowBase + (lane & 15);
  const _Float16* p = Wf + m * ldk + 32 * kt + 16 * (lane >> 4);
  v8h lo = *(const v8h*)p;
  v8h hi = *(const v8h*)(p + 8);
  return __builtin_shufflevector(lo, hi, 0, 1, 2, 3, 4, 5, 6, 7,
                                 8, 9, 10, 11, 12, 13, 14, 15);
}

template <bool TAIL>
__device__ __forceinline__ v16h frag_b_col(const _Float16* __restrict__ Bm, int ldk,
                                           int colBase, int kt, int lane) {
  const int n = colBase + (lane & 15);
  const int hlf = lane >> 4;
  v8h lo = {(_Float16)0, (_Float16)0, (_Float16)0, (_Float16)0,
            (_Float16)0, (_Float16)0, (_Float16)0, (_Float16)0};
  v8h hi = lo;
  if (!TAIL || hlf == 0) {  // TAIL: upper half-wave covers K>=80 (all zero pad)
    const _Float16* p = Bm + n * ldk + 32 * kt + 16 * hlf;
    lo = *(const v8h*)p;
    hi = *(const v8h*)(p + 8);
  }
  return __builtin_shufflevector(lo, hi, 0, 1, 2, 3, 4, 5, 6, 7,
                                 8, 9, 10, 11, 12, 13, 14, 15);
}

__device__ __forceinline__ void load_bias8(const float* __restrict__ b, int base,
                                           float* arr) {
  v4f lo = *(const v4f*)&b[base];
  v4f hi = *(const v4f*)&b[base + 4];
#pragma unroll
  for (int j = 0; j < 4; ++j) { arr[j] = lo[j]; arr[4 + j] = hi[j]; }
}

__device__ __forceinline__ float lrelu01(float v) { return v > 0.0f ? v : 0.1f * v; }
__device__ __forceinline__ float sigm(float v) { return 1.0f / (1.0f + __expf(-v)); }

// ---------------------------------------------------------------------------
// Kernel 2: fused GRU mapping for 8 query points (128 cols, k-minor: each
// 16-col WMMA tile == one query's 16 neighbors, so max-over-k is a 16-lane
// shfl reduction). 4 waves; wave w owns output channel rows [16w,16w+16).
// ---------------------------------------------------------------------------
constexpr int LDX = 80;  // sX column stride in halfs (67 valid, zero to 80)

__global__ __launch_bounds__(128) void gru_kernel(
    const float* __restrict__ xyz1, const float* __restrict__ xyz2,
    const float* __restrict__ points1, const float* __restrict__ points2,
    const float* __restrict__ b_r0, const float* __restrict__ b_r1,
    const float* __restrict__ b_z0, const float* __restrict__ b_z1,
    const float* __restrict__ b_h0, const float* __restrict__ b_h1,
    const _Float16* __restrict__ wf,
    const int* __restrict__ knn, float* __restrict__ out) {
  __shared__ _Float16 sX[128 * LDX];  // new_pts [col][K], reused later as r*p1
  __shared__ _Float16 sT[128 * 64];   // lrelu(r_pre) [col][C]
  __shared__ _Float16 sU[128 * 64];   // raw h0 + bias [col][C]
  __shared__ _Float16 sPt[16 * 64];   // points1 tile [q][C] (q 8..15 zero)
  __shared__ float    sP1f[8 * 64];   // points1 tile f32 [q][C]
  __shared__ _Float16 sVec[16 * 64];  // fuse_r then fuse_z [q][C]
  __shared__ _Float16 sZm[16 * 64];   // max_k z [q][C]
  __shared__ _Float16 sHm[16 * 64];   // max_k h [q][C]
  __shared__ int      sIdx[128];
  __shared__ float    sQ[8 * 3];

  const int tid = threadIdx.x;
  const int lane = tid & 31;
  const int wave = tid >> 5;
  const int rowBase = wave * 16;
  const int hlf = lane >> 4;
  const int mBase = rowBase + 8 * hlf;  // first of 8 consecutive out rows/lane
  const int b = blockIdx.y;
  const int qbase = blockIdx.x * 8;

  // ---- stage 0: knn indices, query coords, points1 tile ----
  {
    const int q = tid >> 4, k = tid & 15;
    sIdx[tid] = knn[((size_t)(b * N1_) + (qbase + q)) * KNN_ + k];
    if (tid < 24) {
      const int q2 = tid / 3, d = tid % 3;
      sQ[q2 * 3 + d] = xyz1[((size_t)b * 3 + d) * N1_ + qbase + q2];
    }
    for (int i = tid; i < 16 * 64; i += 128) {
      const int q2 = i >> 6, c = i & 63;
      float v = 0.0f;
      if (q2 < 8) {
        v = points1[((size_t)b * 64 + c) * N1_ + qbase + q2];
        sP1f[q2 * 64 + c] = v;
      }
      sPt[i] = (_Float16)v;
    }
  }
  __syncthreads();

  // ---- stage 1: gather new_pts columns (p2 gather + dir_xyz + zero pad) ----
  {
    const int col = tid, q = col >> 4;
    const int idx = sIdx[col];
    _Float16* xc = sX + col * LDX;
    const float* p2 = points2 + (size_t)b * 64 * N2_;
    for (int c = 0; c < 64; ++c) xc[c] = (_Float16)p2[c * N2_ + idx];
    const float* x2 = xyz2 + (size_t)b * 3 * N2_;
#pragma unroll
    for (int d = 0; d < 3; ++d)
      xc[64 + d] = (_Float16)(x2[d * N2_ + idx] - sQ[q * 3 + d]);
#pragma unroll
    for (int c = CIN_; c < LDX; ++c) xc[c] = (_Float16)0.0f;
  }
  __syncthreads();

  // ---- stage 2: fr = w_fuse_r x points1_tile ----
  {
    v16h a0 = frag_a_wf(wf + WO_FR, 64, rowBase, 0, lane);
    v16h a1 = frag_a_wf(wf + WO_FR, 64, rowBase, 1, lane);
    v8f acc = {0.f, 0.f, 0.f, 0.f, 0.f, 0.f, 0.f, 0.f};
    WMMA_F16(acc, a0, frag_b_col<false>(sPt, 64, 0, 0, lane));
    WMMA_F16(acc, a1, frag_b_col<false>(sPt, 64, 0, 1, lane));
    v8h pk;
#pragma unroll
    for (int j = 0; j < 8; ++j) pk[j] = (_Float16)acc[j];
    *(v8h*)&sVec[(lane & 15) * 64 + mBase] = pk;
  }
  __syncthreads();

  // ---- stage 3: r_pre = lrelu(w_r0 x X + b_r0 + fr) -> sT ----
  {
    float bias[8];
    load_bias8(b_r0, mBase, bias);
    v16h a0 = frag_a_wf(wf + WO_R0, 96, rowBase, 0, lane);
    v16h a1 = frag_a_wf(wf + WO_R0, 96, rowBase, 1, lane);
    v16h a2 = frag_a_wf(wf + WO_R0, 96, rowBase, 2, lane);
    for (int ct = 0; ct < 8; ++ct) {
      v8f acc = {0.f, 0.f, 0.f, 0.f, 0.f, 0.f, 0.f, 0.f};
      WMMA_F16(acc, a0, frag_b_col<false>(sX, LDX, ct * 16, 0, lane));
      WMMA_F16(acc, a1, frag_b_col<false>(sX, LDX, ct * 16, 1, lane));
      WMMA_F16(acc, a2, frag_b_col<true>(sX, LDX, ct * 16, 2, lane));
      const int col = ct * 16 + (lane & 15);
      v8h fr = *(const v8h*)&sVec[ct * 64 + mBase];
      v8h pk;
#pragma unroll
      for (int j = 0; j < 8; ++j)
        pk[j] = (_Float16)lrelu01(acc[j] + bias[j] + (float)fr[j]);
      *(v8h*)&sT[col * 64 + mBase] = pk;
    }
  }
  __syncthreads();

  // ---- stage 4: fz = w_fuse_z x points1_tile -> sVec (overwrite) ----
  {
    v16h a0 = frag_a_wf(wf + WO_FZ, 64, rowBase, 0, lane);
    v16h a1 = frag_a_wf(wf + WO_FZ, 64, rowBase, 1, lane);
    v8f acc = {0.f, 0.f, 0.f, 0.f, 0.f, 0.f, 0.f, 0.f};
    WMMA_F16(acc, a0, frag_b_col<false>(sPt, 64, 0, 0, lane));
    WMMA_F16(acc, a1, frag_b_col<false>(sPt, 64, 0, 1, lane));
    v8h pk;
#pragma unroll
    for (int j = 0; j < 8; ++j) pk[j] = (_Float16)acc[j];
    *(v8h*)&sVec[(lane & 15) * 64 + mBase] = pk;
  }
  __syncthreads();

  // ---- stage 5: z = lrelu(w_z0 x X + b_z0 + fz), max over k -> sZm ----
  {
    float bias[8];
    load_bias8(b_z0, mBase, bias);
    v16h a0 = frag_a_wf(wf + WO_Z0, 96, rowBase, 0, lane);
    v16h a1 = frag_a_wf(wf + WO_Z0, 96, rowBase, 1, lane);
    v16h a2 = frag_a_wf(wf + WO_Z0, 96, rowBase, 2, lane);
    for (int ct = 0; ct < 8; ++ct) {
      v8f acc = {0.f, 0.f, 0.f, 0.f, 0.f, 0.f, 0.f, 0.f};
      WMMA_F16(acc, a0, frag_b_col<false>(sX, LDX, ct * 16, 0, lane));
      WMMA_F16(acc, a1, frag_b_col<false>(sX, LDX, ct * 16, 1, lane));
      WMMA_F16(acc, a2, frag_b_col<true>(sX, LDX, ct * 16, 2, lane));
      v8h fz = *(const v8h*)&sVec[ct * 64 + mBase];
#pragma unroll
      for (int j = 0; j < 8; ++j) {
        float v = lrelu01(acc[j] + bias[j] + (float)fz[j]);
#pragma unroll
        for (int off = 1; off < 16; off <<= 1)
          v = fmaxf(v, __shfl_xor(v, off, 32));
        if ((lane & 15) == 0) sZm[ct * 64 + mBase + j] = (_Float16)v;
      }
    }
  }
  __syncthreads();

  // ---- stage 6: raw h0 = w_h0 x X + b_h0 -> sU (lrelu deferred) ----
  {
    float bias[8];
    load_bias8(b_h0, mBase, bias);
    v16h a0 = frag_a_wf(wf + WO_H0, 96, rowBase, 0, lane);
    v16h a1 = frag_a_wf(wf + WO_H0, 96, rowBase, 1, lane);
    v16h a2 = frag_a_wf(wf + WO_H0, 96, rowBase, 2, lane);
    for (int ct = 0; ct < 8; ++ct) {
      v8f acc = {0.f, 0.f, 0.f, 0.f, 0.f, 0.f, 0.f, 0.f};
      WMMA_F16(acc, a0, frag_b_col<false>(sX, LDX, ct * 16, 0, lane));
      WMMA_F16(acc, a1, frag_b_col<false>(sX, LDX, ct * 16, 1, lane));
      WMMA_F16(acc, a2, frag_b_col<true>(sX, LDX, ct * 16, 2, lane));
      const int col = ct * 16 + (lane & 15);
      v8h pk;
#pragma unroll
      for (int j = 0; j < 8; ++j) pk[j] = (_Float16)(acc[j] + bias[j]);
      *(v8h*)&sU[col * 64 + mBase] = pk;
    }
  }
  __syncthreads();

  // ---- stage 7: r = sigmoid(w_r1 x sT + b_r1); r*p1 -> sR (reuses sX) ----
  _Float16* sR = sX;  // X fully consumed; reinterpret as [col][64]
  {
    float bias[8];
    load_bias8(b_r1, mBase, bias);
    v16h a0 = frag_a_wf(wf + WO_R1, 64, rowBase, 0, lane);
    v16h a1 = frag_a_wf(wf + WO_R1, 64, rowBase, 1, lane);
    for (int ct = 0; ct < 8; ++ct) {
      v8f acc = {0.f, 0.f, 0.f, 0.f, 0.f, 0.f, 0.f, 0.f};
      WMMA_F16(acc, a0, frag_b_col<false>(sT, 64, ct * 16, 0, lane));
      WMMA_F16(acc, a1, frag_b_col<false>(sT, 64, ct * 16, 1, lane));
      const int col = ct * 16 + (lane & 15);
      v4f p1lo = *(const v4f*)&sP1f[ct * 64 + mBase];
      v4f p1hi = *(const v4f*)&sP1f[ct * 64 + mBase + 4];
      v8h pk;
#pragma unroll
      for (int j = 0; j < 8; ++j) {
        const float r = sigm(acc[j] + bias[j]);
        const float p1 = (j < 4) ? p1lo[j & 3] : p1hi[j & 3];
        pk[j] = (_Float16)(r * p1);
      }
      *(v8h*)&sR[col * 64 + mBase] = pk;
    }
  }
  __syncthreads();

  // ---- stage 8: h = lrelu(h0 + w_fuse_ro x (r*p1)), max over k -> sHm ----
  {
    v16h a0 = frag_a_wf(wf + WO_FRO, 64, rowBase, 0, lane);
    v16h a1 = frag_a_wf(wf + WO_FRO, 64, rowBase, 1, lane);
    for (int ct = 0; ct < 8; ++ct) {
      v8f acc = {0.f, 0.f, 0.f, 0.f, 0.f, 0.f, 0.f, 0.f};
      WMMA_F16(acc, a0, frag_b_col<false>(sR, 64, ct * 16, 0, lane));
      WMMA_F16(acc, a1, frag_b_col<false>(sR, 64, ct * 16, 1, lane));
      const int col = ct * 16 + (lane & 15);
      v8h h0v = *(const v8h*)&sU[col * 64 + mBase];
#pragma unroll
      for (int j = 0; j < 8; ++j) {
        float v = lrelu01(acc[j] + (float)h0v[j]);
#pragma unroll
        for (int off = 1; off < 16; off <<= 1)
          v = fmaxf(v, __shfl_xor(v, off, 32));
        if ((lane & 15) == 0) sHm[ct * 64 + mBase + j] = (_Float16)v;
      }
    }
  }
  __syncthreads();

  // ---- stages 9-11: z1/h1 head GEMMs + final blend (columns = queries) ----
  {
    float bz[8], bh[8];
    load_bias8(b_z1, mBase, bz);
    load_bias8(b_h1, mBase, bh);
    v8f accz = {0.f, 0.f, 0.f, 0.f, 0.f, 0.f, 0.f, 0.f};
    v8f acch = accz;
    WMMA_F16(accz, frag_a_wf(wf + WO_Z1, 64, rowBase, 0, lane),
             frag_b_col<false>(sZm, 64, 0, 0, lane));
    WMMA_F16(accz, frag_a_wf(wf + WO_Z1, 64, rowBase, 1, lane),
             frag_b_col<false>(sZm, 64, 0, 1, lane));
    WMMA_F16(acch, frag_a_wf(wf + WO_H1, 64, rowBase, 0, lane),
             frag_b_col<false>(sHm, 64, 0, 0, lane));
    WMMA_F16(acch, frag_a_wf(wf + WO_H1, 64, rowBase, 1, lane),
             frag_b_col<false>(sHm, 64, 0, 1, lane));
    const int q = lane & 15;
    if (q < 8) {
      v4f p1lo = *(const v4f*)&sP1f[q * 64 + mBase];
      v4f p1hi = *(const v4f*)&sP1f[q * 64 + mBase + 4];
#pragma unroll
      for (int j = 0; j < 8; ++j) {
        const float z = sigm(accz[j] + bz[j]);
        const float h = tanhf(acch[j] + bh[j]);
        const float p1 = (j < 4) ? p1lo[j & 3] : p1hi[j & 3];
        out[((size_t)b * 64 + mBase + j) * N1_ + qbase + q] =
            (1.0f - z) * p1 + z * h;
      }
    }
  }
}

// ---------------------------------------------------------------------------
extern "C" void kernel_launch(void* const* d_in, const int* in_sizes, int n_in,
                              void* d_out, int out_size, void* d_ws, size_t ws_size,
                              hipStream_t stream) {
  (void)in_sizes; (void)n_in; (void)out_size; (void)ws_size;
  const float* xyz1      = (const float*)d_in[0];
  const float* xyz2      = (const float*)d_in[1];
  const float* points1   = (const float*)d_in[2];
  const float* points2   = (const float*)d_in[3];
  const float* w_r0      = (const float*)d_in[4];
  const float* b_r0      = (const float*)d_in[5];
  const float* w_r1      = (const float*)d_in[6];
  const float* b_r1      = (const float*)d_in[7];
  const float* w_z0      = (const float*)d_in[8];
  const float* b_z0      = (const float*)d_in[9];
  const float* w_z1      = (const float*)d_in[10];
  const float* b_z1      = (const float*)d_in[11];
  const float* w_h0      = (const float*)d_in[12];
  const float* b_h0      = (const float*)d_in[13];
  const float* w_h1      = (const float*)d_in[14];
  const float* b_h1      = (const float*)d_in[15];
  const float* w_fuse_r  = (const float*)d_in[16];
  const float* w_fuse_z  = (const float*)d_in[17];
  const float* w_fuse_ro = (const float*)d_in[18];

  int*      knn = (int*)d_ws;                                // 512 KB
  _Float16* wf  = (_Float16*)((char*)d_ws + (size_t)B_ * N1_ * KNN_ * 4);
  float*    out = (float*)d_out;                             // [B, C, N1] f32

  wcvt_kernel<<<(WF_TOTAL + 255) / 256, 256, 0, stream>>>(
      w_r0, w_z0, w_h0, w_r1, w_z1, w_h1, w_fuse_r, w_fuse_z, w_fuse_ro, wf);

  dim3 g1(N1_ / 256, B_);
  knn_kernel<<<g1, 256, 0, stream>>>(xyz1, xyz2, knn);

  dim3 g2(N1_ / 8, B_);
  gru_kernel<<<g2, 128, 0, stream>>>(xyz1, xyz2, points1, points2,
                                     b_r0, b_r1, b_z0, b_z1, b_h0, b_h1,
                                     wf, knn, out);
}
